// NNK_17274358465083
// MI455X (gfx1250) — matrix operand order repaired and checked by reading, not compiled, not run on hardware
//
#include <hip/hip_runtime.h>
#include <hip/hip_bf16.h>

// ---------------------------------------------------------------------------
// Performer random-feature pipeline for MI455X (gfx1250, wave32, WMMA).
//   GEMM1: W  (1024x1024) x rg^T -> weights (exp epilogue, bf16 out)
//   GEMM2: x  (8192x1024) x rg^T -> feats   (exp epilogue, bf16 out)
//   GEMM3: feats (8192x1024) x weights^T -> out (f32)
// NT GEMMs on bf16 operands, f32 accumulation via v_wmma_f32_16x16x32_bf16.
// Global->LDS staging uses gfx1250 async-to-LDS (ASYNCcnt) with a 2-stage
// double-buffered pipeline: wait(stage s) -> barrier -> issue(stage s+1,
// other buffer) -> 8x WMMA on stage s.
// ---------------------------------------------------------------------------

typedef unsigned short u16;
typedef __attribute__((ext_vector_type(16))) __bf16 v16bf;
typedef __attribute__((ext_vector_type(8)))  float  v8f;
typedef int v4i __attribute__((vector_size(4 * sizeof(int))));   // matches builtin

#define AS1 __attribute__((address_space(1)))
#define AS3 __attribute__((address_space(3)))

#define DIMK   1024   // DIM == NUM_RFS == OUT_DIM == 1024
#define MX     8192   // B*S
#define BM     128
#define BN     64
#define BK     64
#define LDA    (BK + 8)   // 72 halves = 144B stride: 16B aligned, conflict-free
#define LDB    (BK + 8)

__device__ __forceinline__ u16 f2bf(float f) {
    unsigned int u = __float_as_uint(f);
    unsigned int r = u + 0x7FFFu + ((u >> 16) & 1u);   // round-to-nearest-even
    return (u16)(r >> 16);
}

// 16B global -> LDS async copy (per-lane), tracked by ASYNCcnt.
__device__ __forceinline__ void async_cp16(const u16* __restrict__ g, u16* l) {
#if __has_builtin(__builtin_amdgcn_global_load_async_to_lds_b128)
    __builtin_amdgcn_global_load_async_to_lds_b128(
        (AS1 v4i*)g, (AS3 v4i*)l, 0, 0);
#else
    unsigned loff = (unsigned)(__UINTPTR_TYPE__)(AS3 u16*)l;
    asm volatile("global_load_async_to_lds_b128 %0, %1, off"
                 :: "v"(loff), "v"(g) : "memory");
#endif
}

__device__ __forceinline__ void wait_async0() {
#if __has_builtin(__builtin_amdgcn_s_wait_asynccnt)
    __builtin_amdgcn_s_wait_asynccnt(0);
#else
    asm volatile("s_wait_asynccnt 0" ::: "memory");
#endif
}

// f32 -> bf16 row convert + 0.5*||row||^2
__global__ __launch_bounds__(256) void row_prep(const float* __restrict__ src,
                                                u16* __restrict__ dst,
                                                float* __restrict__ norm,
                                                int ncols) {
    __shared__ float red[256];
    const int row = blockIdx.x;
    const float* s = src + (size_t)row * ncols;
    u16* d = dst + (size_t)row * ncols;
    float acc = 0.f;
    for (int i = threadIdx.x; i < ncols; i += 256) {
        float v = s[i];
        acc += v * v;
        d[i] = f2bf(v);
    }
    red[threadIdx.x] = acc;
    __syncthreads();
    for (int off = 128; off > 0; off >>= 1) {
        if (threadIdx.x < (unsigned)off) red[threadIdx.x] += red[threadIdx.x + off];
        __syncthreads();
    }
    if (threadIdx.x == 0) norm[row] = 0.5f * red[0];
}

// rg[r][d] = bf16(xis[r] * proj[r][d]); 4 elements/thread
__global__ __launch_bounds__(256) void rg_prep(const float* __restrict__ proj,
                                               const float* __restrict__ xis,
                                               u16* __restrict__ rg) {
    size_t idx = ((size_t)blockIdx.x * 256 + threadIdx.x) * 4;
    int r = (int)(idx >> 10);                 // / DIMK
    float xi = xis[r];
    float4 p = *reinterpret_cast<const float4*>(proj + idx);
    unsigned long long pack =
          (unsigned long long)f2bf(xi * p.x)
        | ((unsigned long long)f2bf(xi * p.y) << 16)
        | ((unsigned long long)f2bf(xi * p.z) << 32)
        | ((unsigned long long)f2bf(xi * p.w) << 48);
    *reinterpret_cast<unsigned long long*>(rg + idx) = pack;
}

// NT GEMM: out = A[M][K] * B[N][K]^T, bf16 in, f32 accumulate.
// EPI=true : out_bf[m][n] = bf16( exp(acc + xi[n] - norms[m]*xi[n]^2) / 32 )
// EPI=false: out_f32[m][n] = acc
template <bool EPI>
__global__ __launch_bounds__(256) void rfs_gemm(const u16* __restrict__ A,
                                                const u16* __restrict__ B,
                                                const float* __restrict__ norms,
                                                const float* __restrict__ xis,
                                                u16* __restrict__ outb,
                                                float* __restrict__ outf,
                                                int M, int N, int K) {
    __shared__ u16 lsA[2][BM * LDA];
    __shared__ u16 lsB[2][BN * LDB];

    const int tid  = threadIdx.x;
    const int lane = tid & 31;
    const int wave = tid >> 5;
    const int ln15 = lane & 15;
    const int hi   = lane >> 4;          // half-wave select
    const int wm   = wave & 3;           // 4 wave-rows  x 32
    const int wn   = wave >> 2;          // 2 wave-cols  x 32
    const int blockM = blockIdx.y * BM;
    const int blockN = blockIdx.x * BN;

    v8f c00 = {}, c01 = {}, c10 = {}, c11 = {};

    // async stage fill: A tile 128x64 halves = 1024 x16B chunks (4/thread),
    //                   B tile  64x64 halves =  512 x16B chunks (2/thread)
    auto issue_stage = [&](int buf, int k0) {
        #pragma unroll
        for (int i = 0; i < 4; ++i) {
            int c   = tid + i * 256;
            int row = c >> 3;            // 8 chunks per 64-half row
            int cc  = c & 7;
            async_cp16(A + (size_t)(blockM + row) * K + k0 + cc * 8,
                       &lsA[buf][row * LDA + cc * 8]);
        }
        #pragma unroll
        for (int i = 0; i < 2; ++i) {
            int c   = tid + i * 256;
            int row = c >> 3;
            int cc  = c & 7;
            async_cp16(B + (size_t)(blockN + row) * K + k0 + cc * 8,
                       &lsB[buf][row * LDB + cc * 8]);
        }
    };

    const int nstage = K / BK;
    issue_stage(0, 0);

    for (int s = 0; s < nstage; ++s) {
        wait_async0();        // this wave's stage-s writes have landed
        __syncthreads();      // everyone's stage-s writes visible; prev reads done
        if (s + 1 < nstage) issue_stage((s + 1) & 1, (s + 1) * BK);

        const int cur = s & 1;
        #pragma unroll
        for (int ks = 0; ks < 2; ++ks) {         // two 16x16x32 K-steps
            const int koff = ks * 32;
            // per-lane fragments per ISA 7.12.2:
            // A 16x32 bf16: row = lane&15; halves 0..7 -> K=8*hi+0..7,
            //               halves 8..15 -> K=16+8*hi+0..7
            union Frag { uint4 q[2]; v16bf v; };
            Frag a0, a1, b0, b1;
            const u16* pa0 = &lsA[cur][(wm * 32 + ln15) * LDA + koff + hi * 8];
            const u16* pa1 = pa0 + 16 * LDA;
            a0.q[0] = *reinterpret_cast<const uint4*>(pa0);
            a0.q[1] = *reinterpret_cast<const uint4*>(pa0 + 16);
            a1.q[0] = *reinterpret_cast<const uint4*>(pa1);
            a1.q[1] = *reinterpret_cast<const uint4*>(pa1 + 16);
            // B 32x16 bf16: col = lane&15; halves 0..7 -> K=8*hi+0..7,
            //               halves 8..15 -> K=16+8*hi+0..7
            const u16* pb0 = &lsB[cur][(wn * 32 + ln15) * LDB + koff + hi * 8];
            const u16* pb1 = pb0 + 16 * LDB;
            b0.q[0] = *reinterpret_cast<const uint4*>(pb0);
            b0.q[1] = *reinterpret_cast<const uint4*>(pb0 + 16);
            b1.q[0] = *reinterpret_cast<const uint4*>(pb1);
            b1.q[1] = *reinterpret_cast<const uint4*>(pb1 + 16);

            c00 = __builtin_amdgcn_wmma_f32_16x16x32_bf16(false, a0.v, false, b0.v,
                                                          (short)0, c00, false, false);
            c01 = __builtin_amdgcn_wmma_f32_16x16x32_bf16(false, a0.v, false, b1.v,
                                                          (short)0, c01, false, false);
            c10 = __builtin_amdgcn_wmma_f32_16x16x32_bf16(false, a1.v, false, b0.v,
                                                          (short)0, c10, false, false);
            c11 = __builtin_amdgcn_wmma_f32_16x16x32_bf16(false, a1.v, false, b1.v,
                                                          (short)0, c11, false, false);
        }
    }

    // D 16x16 f32: lane col n = lane&15; vgpr v -> row m = v + 8*hi
    auto store = [&](v8f c, int fm, int fn) {
        int n     = blockN + wn * 32 + fn * 16 + ln15;
        int mbase = blockM + wm * 32 + fm * 16 + hi * 8;
        if (EPI) {
            float xi  = xis[n];
            float xi2 = xi * xi;
            #pragma unroll
            for (int v = 0; v < 8; ++v) {
                int m = mbase + v;
                float val = expf(c[v] + xi - norms[m] * xi2) * 0.03125f; // 1/sqrt(1024)
                outb[(size_t)m * N + n] = f2bf(val);
            }
        } else {
            #pragma unroll
            for (int v = 0; v < 8; ++v)
                outf[(size_t)(mbase + v) * N + n] = c[v];
        }
    };
    store(c00, 0, 0);
    store(c01, 0, 1);
    store(c10, 1, 0);
    store(c11, 1, 1);
}

extern "C" void kernel_launch(void* const* d_in, const int* in_sizes, int n_in,
                              void* d_out, int out_size, void* d_ws, size_t ws_size,
                              hipStream_t stream) {
    const float* x    = (const float*)d_in[0];   // [4,2048,1024]
    const float* W    = (const float*)d_in[1];   // [1024,1024]
    const float* xis  = (const float*)d_in[2];   // [1024]
    const float* proj = (const float*)d_in[3];   // [1024,1024]
    float* out = (float*)d_out;                  // [4,2048,1024]

    // workspace layout (~40 MB)
    u16* xbf   = (u16*)d_ws;                       // 8192*1024 bf16
    u16* wbf   = xbf   + (size_t)MX * DIMK;        // 1024*1024
    u16* rg    = wbf   + (size_t)DIMK * DIMK;      // 1024*1024
    u16* featb = rg    + (size_t)DIMK * DIMK;      // 8192*1024
    u16* wOut  = featb + (size_t)MX * DIMK;        // 1024*1024
    float* xnorm = (float*)(wOut + (size_t)DIMK * DIMK);
    float* wnorm = xnorm + MX;

    row_prep<<<MX,   256, 0, stream>>>(x, xbf, xnorm, DIMK);
    row_prep<<<DIMK, 256, 0, stream>>>(W, wbf, wnorm, DIMK);
    rg_prep<<<(DIMK * DIMK) / (256 * 4), 256, 0, stream>>>(proj, xis, rg);

    // GEMM1: weights[o][r]  (bf16, exp epilogue)
    rfs_gemm<true><<<dim3(DIMK / BN, DIMK / BM), 256, 0, stream>>>(
        wbf, rg, wnorm, xis, wOut, nullptr, DIMK, DIMK, DIMK);
    // GEMM2: feats[m][r]    (bf16, exp epilogue)
    rfs_gemm<true><<<dim3(DIMK / BN, MX / BM), 256, 0, stream>>>(
        xbf, rg, xnorm, xis, featb, nullptr, MX, DIMK, DIMK);
    // GEMM3: out[m][o] = feats . weights^T (f32)
    rfs_gemm<false><<<dim3(DIMK / BN, MX / BM), 256, 0, stream>>>(
        featb, wOut, nullptr, nullptr, nullptr, out, MX, DIMK, DIMK);
}
